// BlockMulti_44186623541662
// MI455X (gfx1250) — matrix operand order repaired and checked by reading, not compiled, not run on hardware
//
#include <hip/hip_runtime.h>
#include <hip/hip_bf16.h>
#include <math.h>

#define B_   4
#define S_   2048
#define D_   1024
#define H_   16
#define DH_  64
#define MLP_ 4096
#define T_   (B_*S_)

typedef __attribute__((ext_vector_type(16))) _Float16 v16h;
typedef __attribute__((ext_vector_type(8)))  _Float16 v8h;
typedef __attribute__((ext_vector_type(8)))  float    v8f;
typedef int v4i_vs __attribute__((vector_size(4 * sizeof(int))));

#define WMMA_F16(a,b,c) __builtin_amdgcn_wmma_f32_16x16x32_f16(false,(a),false,(b),(short)0,(c),false,false)

// ---------------- CDNA5 async global->LDS copy (ASYNCcnt) ----------------
#if defined(__has_builtin)
#if __has_builtin(__builtin_amdgcn_global_load_async_to_lds_b128)
#define HAVE_ASYNC_BUILTIN 1
#endif
#if __has_builtin(__builtin_amdgcn_s_wait_asynccnt)
#define HAVE_ASYNCWAIT_BUILTIN 1
#endif
#endif

static __device__ __forceinline__ void async_b128(const _Float16* g, _Float16* l) {
#if defined(HAVE_ASYNC_BUILTIN) && defined(__AMDGCN__)
  __builtin_amdgcn_global_load_async_to_lds_b128(
      (__attribute__((address_space(1))) v4i_vs*)g,
      (__attribute__((address_space(3))) v4i_vs*)l, 0, 0);
#else
  // vdst = LDS byte address (low 32 bits of flat shared pointer), vaddr = global addr
  asm volatile("global_load_async_to_lds_b128 %0, %1, off"
               :: "v"((unsigned)(unsigned long long)l), "v"(g)
               : "memory");
#endif
}
template<int N>
static __device__ __forceinline__ void wait_async() {
#if defined(HAVE_ASYNCWAIT_BUILTIN) && defined(__AMDGCN__)
  __builtin_amdgcn_s_wait_asynccnt(N);
#else
  asm volatile("s_wait_asynccnt %0" :: "n"(N) : "memory");
#endif
}

// ---------------- WMMA fragment loaders (16x16x32 f16, wave32) ----------------
// A 16x32 (MxK): lane<16: M=lane, K=0..15; lane>=16: M=lane-16, K=16..31
static __device__ __forceinline__ v16h ld_frag_a(const _Float16* p, int ld) {
  const int lane = threadIdx.x & 31;
  const int m  = lane & 15;
  const int kh = (lane & 16) ? 16 : 0;
  const _Float16* r = p + m * ld + kh;
  v16h a;
#pragma unroll
  for (int i = 0; i < 8; ++i) { a[i] = r[i]; a[i + 8] = r[8 + i]; }
  return a;
}
// B 32x16 (KxN) from N-major storage (row n holds K contiguous): contiguous 16-half read
static __device__ __forceinline__ v16h ld_frag_bt(const _Float16* p, int ld) {
  const int lane = threadIdx.x & 31;
  const int n  = lane & 15;
  const int kb = (lane & 16) ? 16 : 0;
  const _Float16* r = p + n * ld + kb;
  v16h b;
#pragma unroll
  for (int i = 0; i < 16; ++i) b[i] = r[i];
  return b;
}

// ---------------- weight fp32 -> fp16 with transpose (KxN -> NxK) ----------------
__global__ __launch_bounds__(256) void f32_to_f16_T(
    const float* __restrict__ s, _Float16* __restrict__ d, int K, int N)
{
  __shared__ _Float16 t[32][33];
  const int k0 = blockIdx.x * 32, n0 = blockIdx.y * 32;
  const int tx = threadIdx.x & 31, ty = threadIdx.x >> 5;  // ty 0..7
#pragma unroll
  for (int yy = ty; yy < 32; yy += 8)
    t[yy][tx] = (_Float16)s[(size_t)(k0 + yy) * N + n0 + tx];
  __syncthreads();
#pragma unroll
  for (int yy = ty; yy < 32; yy += 8)
    d[(size_t)(n0 + yy) * K + k0 + tx] = t[tx][yy];
}

// ---------------- LayerNorm: fp32 in -> f16 normalized out ----------------
__global__ __launch_bounds__(256) void layernorm_f16(
    const float* __restrict__ x, const float* __restrict__ g,
    const float* __restrict__ bta, _Float16* __restrict__ o, int cols)
{
  __shared__ float sred[2][8];
  const int row = blockIdx.x, tid = threadIdx.x, lane = tid & 31, wv = tid >> 5;
  const float* xr = x + (size_t)row * cols;
  float s = 0.f, ss = 0.f;
  for (int c = tid; c < cols; c += 256) { float t = xr[c]; s += t; ss += t * t; }
#pragma unroll
  for (int msk = 16; msk >= 1; msk >>= 1) {
    s  += __shfl_xor(s,  msk, 32);
    ss += __shfl_xor(ss, msk, 32);
  }
  if (lane == 0) { sred[0][wv] = s; sred[1][wv] = ss; }
  __syncthreads();
  s = 0.f; ss = 0.f;
#pragma unroll
  for (int i = 0; i < 8; ++i) { s += sred[0][i]; ss += sred[1][i]; }
  const float mu = s / cols;
  const float var = ss / cols - mu * mu;
  const float rstd = rsqrtf(var + 1e-6f);
  _Float16* orow = o + (size_t)row * cols;
  for (int c = tid; c < cols; c += 256)
    orow[c] = (_Float16)((xr[c] - mu) * rstd * g[c] + bta[c]);
}

// ---------------- WMMA GEMM: C = A(f16,MxK) x B^T(f16,NxK) + bias ----------------
// 128x128 block tile, BK=32, double-buffered async LDS staging, 8 WMMAs/wave/step.
// EPI 0: bias -> f16 ; EPI 1: bias + resid -> f32 ; EPI 2: bias + exact GELU -> f16
template<int EPI>
__global__ __launch_bounds__(256) void gemm_wmma(
    const _Float16* __restrict__ A, const _Float16* __restrict__ Bt,
    const float* __restrict__ bias, const float* __restrict__ resid,
    void* __restrict__ outp, int M, int N, int K)
{
  __shared__ __align__(16) _Float16 sA[2][128 * 32];
  __shared__ __align__(16) _Float16 sB[2][128 * 32];   // N-major: row n, 32 K-halfs
  const int tid = threadIdx.x, lane = tid & 31, wv = tid >> 5;
  const int bm = blockIdx.x * 128, bn = blockIdx.y * 128;
  const int wm = (wv & 3) * 32, wn = (wv >> 2) * 64;
  v8f acc[2][4] = {};

  // staging: thread t covers rows r and r+64, 8-half column chunk
  const int r = tid >> 2, cchunk = (tid & 3) * 8;
  const _Float16* gA0 = A  + (size_t)(bm + r)      * K + cchunk;
  const _Float16* gA1 = A  + (size_t)(bm + r + 64) * K + cchunk;
  const _Float16* gB0 = Bt + (size_t)(bn + r)      * K + cchunk;
  const _Float16* gB1 = Bt + (size_t)(bn + r + 64) * K + cchunk;
  _Float16* lA0 = &sA[0][r * 32 + cchunk];
  _Float16* lA1 = &sA[0][(r + 64) * 32 + cchunk];
  _Float16* lB0 = &sB[0][r * 32 + cchunk];
  _Float16* lB1 = &sB[0][(r + 64) * 32 + cchunk];
  const size_t bufStride = 128 * 32;

  const int nk = K / 32;
  // prologue: stage k-step 0 into buffer 0
  async_b128(gA0, lA0); async_b128(gA1, lA1);
  async_b128(gB0, lB0); async_b128(gB1, lB1);

  for (int i = 0; i < nk; ++i) {
    const int buf = i & 1;
    if (i + 1 < nk) {
      const size_t ko = (size_t)(i + 1) * 32, bo = (size_t)((i + 1) & 1) * bufStride;
      async_b128(gA0 + ko, lA0 + bo); async_b128(gA1 + ko, lA1 + bo);
      async_b128(gB0 + ko, lB0 + bo); async_b128(gB1 + ko, lB1 + bo);
      wait_async<4>();
    } else {
      wait_async<0>();
    }
    __syncthreads();  // this k-step's tiles visible to all waves
    const _Float16* cA = sA[buf];
    const _Float16* cB = sB[buf];
    v16h a0 = ld_frag_a(cA + (size_t)wm * 32, 32);
    v16h a1 = ld_frag_a(cA + (size_t)(wm + 16) * 32, 32);
    v16h b0 = ld_frag_bt(cB + (size_t)(wn +  0) * 32, 32);
    v16h b1 = ld_frag_bt(cB + (size_t)(wn + 16) * 32, 32);
    v16h b2 = ld_frag_bt(cB + (size_t)(wn + 32) * 32, 32);
    v16h b3 = ld_frag_bt(cB + (size_t)(wn + 48) * 32, 32);
    acc[0][0] = WMMA_F16(a0, b0, acc[0][0]);
    acc[0][1] = WMMA_F16(a0, b1, acc[0][1]);
    acc[0][2] = WMMA_F16(a0, b2, acc[0][2]);
    acc[0][3] = WMMA_F16(a0, b3, acc[0][3]);
    acc[1][0] = WMMA_F16(a1, b0, acc[1][0]);
    acc[1][1] = WMMA_F16(a1, b1, acc[1][1]);
    acc[1][2] = WMMA_F16(a1, b2, acc[1][2]);
    acc[1][3] = WMMA_F16(a1, b3, acc[1][3]);
    __syncthreads();  // all waves done reading buf before it is restaged
  }

  const int n_in = lane & 15, mb = (lane & 16) ? 8 : 0;
#pragma unroll
  for (int i = 0; i < 2; ++i)
#pragma unroll
  for (int j = 0; j < 4; ++j) {
    const int gn = bn + wn + 16 * j + n_in;
    const float bb = bias[gn];
#pragma unroll
    for (int jj = 0; jj < 8; ++jj) {
      const int gm = bm + wm + 16 * i + mb + jj;
      float vv = acc[i][j][jj] + bb;
      if constexpr (EPI == 2) vv = 0.5f * vv * (1.0f + erff(vv * 0.7071067811865475f));
      const size_t idx = (size_t)gm * N + gn;
      if constexpr (EPI == 1) ((float*)outp)[idx] = vv + resid[idx];
      else                    ((_Float16*)outp)[idx] = (_Float16)vv;
    }
  }
}

// ---------------- Flash attention: block = (b,h,64 queries), 4 waves ----------------
__global__ __launch_bounds__(128) void flash_attn(
    const _Float16* __restrict__ q, const _Float16* __restrict__ k,
    const _Float16* __restrict__ v, _Float16* __restrict__ ctx)
{
  __shared__ __align__(16) _Float16 sK[2][32 * 64];
  __shared__ __align__(16) _Float16 sV[2][32 * 64];
  __shared__ __align__(16) _Float16 sP[4][16 * 32];  // per-wave P tile
  const int tid = threadIdx.x, lane = tid & 31, wv = tid >> 5;
  const int b = blockIdx.z, h = blockIdx.y, q0 = blockIdx.x * 64;
  const size_t base = ((size_t)b * S_) * D_ + (size_t)h * DH_;
  const int m = lane & 15;
  const int kh = (lane & 16) ? 16 : 0;
  const int n = lane & 15;
  const int kb = (lane & 16) ? 16 : 0;
  const int mb = (lane & 16) ? 8 : 0;

  // Q fragments (pre-scaled by 1/sqrt(DH))
  v16h A0, A1;
  {
    const _Float16* qr = q + base + (size_t)(q0 + wv * 16 + m) * D_;
#pragma unroll
    for (int i = 0; i < 8; ++i) {
      A0[i]     = (_Float16)((float)qr[kh + i]           * 0.125f);
      A0[i + 8] = (_Float16)((float)qr[kh + 8 + i]       * 0.125f);
      A1[i]     = (_Float16)((float)qr[32 + kh + i]      * 0.125f);
      A1[i + 8] = (_Float16)((float)qr[32 + kh + 8 + i]  * 0.125f);
    }
  }
  v16h ones;
#pragma unroll
  for (int i = 0; i < 16; ++i) ones[i] = (_Float16)1.0f;

  float mrow[8], lrow[8];
  v8f c0 = {}, c1 = {}, c2 = {}, c3 = {};
#pragma unroll
  for (int j = 0; j < 8; ++j) { mrow[j] = -INFINITY; lrow[j] = 0.f; }

  // staging: 128 threads, each thread 2x b128 of K and of V
  const int rs = tid >> 2, cs = (tid & 3) * 16;
  const _Float16* gk = k + base + (size_t)rs * D_ + cs;
  const _Float16* gv = v + base + (size_t)rs * D_ + cs;
  _Float16* lk = &sK[0][rs * 64 + cs];
  _Float16* lv = &sV[0][rs * 64 + cs];
  const size_t bufStride = 32 * 64;
  const size_t gStep = (size_t)32 * D_;

  async_b128(gk, lk); async_b128(gk + 8, lk + 8);
  async_b128(gv, lv); async_b128(gv + 8, lv + 8);

  const int nchunks = S_ / 32;
  for (int i = 0; i < nchunks; ++i) {
    const int buf = i & 1;
    if (i + 1 < nchunks) {
      const size_t go = (size_t)(i + 1) * gStep, bo = (size_t)((i + 1) & 1) * bufStride;
      async_b128(gk + go,     lk + bo);
      async_b128(gk + go + 8, lk + bo + 8);
      async_b128(gv + go,     lv + bo);
      async_b128(gv + go + 8, lv + bo + 8);
      wait_async<4>();
    } else {
      wait_async<0>();
    }
    __syncthreads();  // chunk i K/V visible

    const _Float16* cK = sK[buf];
    const _Float16* cV = sV[buf];

    // scores: Q(16x64) x K^T(64x32) — B frag element [d][key] = cK[key*64+d] (contiguous)
    v16h B00, B01, B10, B11;
#pragma unroll
    for (int i2 = 0; i2 < 16; ++i2) {
      B00[i2] = cK[n * 64 + kb + i2];
      B01[i2] = cK[n * 64 + 32 + kb + i2];
      B10[i2] = cK[(16 + n) * 64 + kb + i2];
      B11[i2] = cK[(16 + n) * 64 + 32 + kb + i2];
    }
    v8f s0 = {}, s1 = {};
    s0 = WMMA_F16(A0, B00, s0); s0 = WMMA_F16(A1, B01, s0);
    s1 = WMMA_F16(A0, B10, s1); s1 = WMMA_F16(A1, B11, s1);

    // online softmax: row max via 16-lane butterfly
    float rmax[8];
#pragma unroll
    for (int j = 0; j < 8; ++j) rmax[j] = fmaxf(s0[j], s1[j]);
#pragma unroll
    for (int msk = 1; msk < 16; msk <<= 1)
#pragma unroll
      for (int j = 0; j < 8; ++j) rmax[j] = fmaxf(rmax[j], __shfl_xor(rmax[j], msk, 32));

    float corr[8], p0[8], p1[8];
#pragma unroll
    for (int j = 0; j < 8; ++j) {
      const float mn = fmaxf(mrow[j], rmax[j]);
      corr[j] = __expf(mrow[j] - mn);
      mrow[j] = mn;
      p0[j] = __expf(s0[j] - mn);
      p1[j] = __expf(s1[j] - mn);
      c0[j] *= corr[j]; c1[j] *= corr[j]; c2[j] *= corr[j]; c3[j] *= corr[j];
    }

    // P tile (C-layout) -> wave-private LDS -> A-layout; same-wave DS ops are in order
#pragma unroll
    for (int j = 0; j < 8; ++j) {
      sP[wv][(mb + j) * 32 + n]      = (_Float16)p0[j];
      sP[wv][(mb + j) * 32 + 16 + n] = (_Float16)p1[j];
    }
    __builtin_amdgcn_wave_barrier();
    v16h Ap = ld_frag_a(&sP[wv][0], 32);

    // row sums of P via WMMA with all-ones B: every lane gets its row's sum
    v8f zero = {};
    v8f sums = WMMA_F16(Ap, ones, zero);
#pragma unroll
    for (int j = 0; j < 8; ++j) lrow[j] = lrow[j] * corr[j] + sums[j];

    // ctx += P x V
    v16h Bv0, Bv1, Bv2, Bv3;
#pragma unroll
    for (int i2 = 0; i2 < 16; ++i2) {
      Bv0[i2] = cV[(kb + i2) * 64 + 0 * 16 + n];
      Bv1[i2] = cV[(kb + i2) * 64 + 1 * 16 + n];
      Bv2[i2] = cV[(kb + i2) * 64 + 2 * 16 + n];
      Bv3[i2] = cV[(kb + i2) * 64 + 3 * 16 + n];
    }
    c0 = WMMA_F16(Ap, Bv0, c0); c1 = WMMA_F16(Ap, Bv1, c1);
    c2 = WMMA_F16(Ap, Bv2, c2); c3 = WMMA_F16(Ap, Bv3, c3);

    __syncthreads();  // all waves done with buf before restage
  }

  // finalize: ctx = acc / l, store f16 [B,S,D]
  _Float16* co = ctx + base + (size_t)(q0 + wv * 16) * D_;
#pragma unroll
  for (int j = 0; j < 8; ++j) {
    const float inv = 1.0f / lrow[j];
    const size_t row = (size_t)(mb + j) * D_;
    co[row + 0 * 16 + n] = (_Float16)(c0[j] * inv);
    co[row + 1 * 16 + n] = (_Float16)(c1[j] * inv);
    co[row + 2 * 16 + n] = (_Float16)(c2[j] * inv);
    co[row + 3 * 16 + n] = (_Float16)(c3[j] * inv);
  }
}

extern "C" void kernel_launch(void* const* d_in, const int* in_sizes, int n_in,
                              void* d_out, int out_size, void* d_ws, size_t ws_size,
                              hipStream_t stream)
{
  const float* x    = (const float*)d_in[0];
  const float* ln1g = (const float*)d_in[1];
  const float* ln1b = (const float*)d_in[2];
  const float* Wq   = (const float*)d_in[3];
  const float* bq   = (const float*)d_in[4];
  const float* Wk   = (const float*)d_in[5];
  const float* bk   = (const float*)d_in[6];
  const float* Wv   = (const float*)d_in[7];
  const float* bv   = (const float*)d_in[8];
  const float* Wo   = (const float*)d_in[9];
  const float* bo   = (const float*)d_in[10];
  const float* ln2g = (const float*)d_in[11];
  const float* ln2b = (const float*)d_in[12];
  const float* W1   = (const float*)d_in[13];
  const float* b1   = (const float*)d_in[14];
  const float* W2   = (const float*)d_in[15];
  const float* b2   = (const float*)d_in[16];

  char* ws = (char*)d_ws;
  size_t off = 0;
  auto take = [&](size_t bytes) -> char* {
    char* p = ws + off; off += (bytes + 255) & ~(size_t)255; return p;
  };
  _Float16* wq16  = (_Float16*)take((size_t)D_ * D_ * 2);    // transposed (NxK)
  _Float16* wk16  = (_Float16*)take((size_t)D_ * D_ * 2);
  _Float16* wv16  = (_Float16*)take((size_t)D_ * D_ * 2);
  _Float16* wo16  = (_Float16*)take((size_t)D_ * D_ * 2);
  _Float16* w116  = (_Float16*)take((size_t)D_ * MLP_ * 2);  // MLPxD
  _Float16* w216  = (_Float16*)take((size_t)MLP_ * D_ * 2);  // DxMLP
  float*    x1    = (float*)   take((size_t)T_ * D_ * 4);
  _Float16* xn16  = (_Float16*)take((size_t)T_ * D_ * 2);
  _Float16* q16   = (_Float16*)take((size_t)T_ * D_ * 2);
  _Float16* k16   = (_Float16*)take((size_t)T_ * D_ * 2);
  _Float16* v16   = (_Float16*)take((size_t)T_ * D_ * 2);
  _Float16* ctx16 = (_Float16*)take((size_t)T_ * D_ * 2);
  _Float16* xn2   = xn16;              // xn16 dead after QKV GEMMs
  _Float16* y1    = q16;               // q/k/v/ctx region (64MB) dead after Wo GEMM

  // 1) weights fp32 -> fp16, transposed to N-major for contiguous B fragments
  dim3 tg(D_ / 32, D_ / 32);
  f32_to_f16_T<<<tg, 256, 0, stream>>>(Wq, wq16, D_, D_);
  f32_to_f16_T<<<tg, 256, 0, stream>>>(Wk, wk16, D_, D_);
  f32_to_f16_T<<<tg, 256, 0, stream>>>(Wv, wv16, D_, D_);
  f32_to_f16_T<<<tg, 256, 0, stream>>>(Wo, wo16, D_, D_);
  f32_to_f16_T<<<dim3(D_ / 32, MLP_ / 32), 256, 0, stream>>>(W1, w116, D_, MLP_);
  f32_to_f16_T<<<dim3(MLP_ / 32, D_ / 32), 256, 0, stream>>>(W2, w216, MLP_, D_);

  // 2) LN1
  layernorm_f16<<<T_, 256, 0, stream>>>(x, ln1g, ln1b, xn16, D_);

  // 3) QKV projections (bias -> f16)
  dim3 g1(T_ / 128, D_ / 128);
  gemm_wmma<0><<<g1, 256, 0, stream>>>(xn16, wq16, bq, nullptr, q16, T_, D_, D_);
  gemm_wmma<0><<<g1, 256, 0, stream>>>(xn16, wk16, bk, nullptr, k16, T_, D_, D_);
  gemm_wmma<0><<<g1, 256, 0, stream>>>(xn16, wv16, bv, nullptr, v16, T_, D_, D_);

  // 4) flash attention
  dim3 ga(S_ / 64, H_, B_);
  flash_attn<<<ga, 128, 0, stream>>>(q16, k16, v16, ctx16);

  // 5) output projection + residual (fp32)
  gemm_wmma<1><<<g1, 256, 0, stream>>>(ctx16, wo16, bo, x, x1, T_, D_, D_);

  // 6) LN2
  layernorm_f16<<<T_, 256, 0, stream>>>(x1, ln2g, ln2b, xn2, D_);

  // 7) MLP up + exact GELU (f16)
  gemm_wmma<2><<<dim3(T_ / 128, MLP_ / 128), 256, 0, stream>>>(xn2, w116, b1, nullptr, y1, T_, MLP_, D_);

  // 8) MLP down + residual -> fp32 output
  gemm_wmma<1><<<g1, 256, 0, stream>>>(y1, w216, b2, x1, (float*)d_out, T_, D_, MLP_);
}
